// MDTAWithStrategy_38749194944690
// MI455X (gfx1250) — matrix operand order repaired and checked by reading, not compiled, not run on hardware
//
#include <hip/hip_runtime.h>

typedef __bf16 bf16;
typedef __attribute__((ext_vector_type(16))) __bf16 v16bf;
typedef __attribute__((ext_vector_type(8)))  float  v8f;

constexpr int B_ = 4, N_ = 2304, K_ = 8, C_ = 512, H_ = 8, D_ = 64;
constexpr int NT   = N_ + K_;     // 2312 tokens
constexpr int ROWP = 2336;        // padded rows: 73*32 = 146*16
constexpr int TROW = ROWP / 16;   // 146 row tiles
constexpr int QTR  = 145;         // q tiles covering all rows < NT

// ---------------- WMMA fragment helpers (CDNA5 16x16x32 bf16, wave32) --------
// A (16x32): lane l -> M=l%16; halves -> K = i + (l>=16 ? 16 : 0).
// B (32x16): lane l -> N=l%16; halves -> K = i + (l>=16 ? 16 : 0).
// C/D (16x16 f32): lane l, vgpr r -> M = r + (l>=16 ? 8 : 0), N = l%16.

__device__ __forceinline__ v8f wmma_bf(v16bf a, v16bf b, v8f c) {
  return __builtin_amdgcn_wmma_f32_16x16x32_bf16(false, a, false, b, (short)0, c,
                                                 false, false);
}

// A fragment: rowptr = this lane's row base (row-major bf16), contiguous K
__device__ __forceinline__ v16bf loadA_bf(const bf16* rowptr, int K0, int lane) {
  const bf16* p = rowptr + K0 + ((lane >= 16) ? 16 : 0);
  v16bf a;
#pragma unroll
  for (int i = 0; i < 16; ++i) a[i] = p[i];
  return a;
}

// B fragment from an N-major source (src[n][ldk], K contiguous):
//   p(lane) = base + (n0 + lane%16)*ldk + K0 + (lane>=16 ? 16 : 0), 16 contiguous
__device__ __forceinline__ v16bf loadB_nk(const bf16* base, int ldk, int n0,
                                          int K0, int lane) {
  const bf16* p = base + (size_t)(n0 + (lane & 15)) * ldk + K0 +
                  ((lane >= 16) ? 16 : 0);
  v16bf b;
#pragma unroll
  for (int i = 0; i < 16; ++i) b[i] = p[i];
  return b;
}

// ---------------- Prologue: bf16 conversion / transposition ------------------
// X_in = concat(X, S) -> bf16 [B][ROWP][C], zero-padded rows
__global__ __launch_bounds__(256) void cvt_x_kernel(const float* __restrict__ X,
                                                    const float* __restrict__ S,
                                                    bf16* __restrict__ Xb) {
  size_t i = ((size_t)blockIdx.x * blockDim.x + threadIdx.x) * 4;
  size_t b   = i / ((size_t)ROWP * C_);
  size_t rem = i % ((size_t)ROWP * C_);
  int row = (int)(rem / C_);
  int c   = (int)(rem % C_);
  float4 v = {0.f, 0.f, 0.f, 0.f};
  if (row < N_)      v = *(const float4*)(X + ((size_t)b * N_ + row) * C_ + c);
  else if (row < NT) v = *(const float4*)(S + ((size_t)b * K_ + (row - N_)) * C_ + c);
  bf16* d = Xb + i;
  d[0] = (bf16)v.x; d[1] = (bf16)v.y; d[2] = (bf16)v.z; d[3] = (bf16)v.w;
}

// Wt[n][k] = (bf16)W[k][n]  (consecutive threads along k -> coalesced stores)
__global__ __launch_bounds__(256) void transpose_w_kernel(const float* __restrict__ W,
                                                          bf16* __restrict__ Wt,
                                                          int Krows, int Ncols) {
  size_t i = (size_t)blockIdx.x * blockDim.x + threadIdx.x;
  int k = (int)(i % Krows);
  int n = (int)(i / Krows);
  Wt[(size_t)n * Krows + k] = (bf16)W[(size_t)k * Ncols + n];
}

// ---------------- Kernel 1: fused QKV projection + q/k L2-normalize ----------
// q,k stored [B][H][row][64]; v stored transposed [B][H][64][row] for P@V
__global__ __launch_bounds__(32) void qkv_kernel(const bf16* __restrict__ Xb,
                                                 const bf16* __restrict__ Wqkvt,
                                                 bf16* __restrict__ qb,
                                                 bf16* __restrict__ kb,
                                                 bf16* __restrict__ vT) {
  const int lane = threadIdx.x;
  int idx = blockIdx.x;
  const int b  = idx / (TROW * 24);
  const int r  = idx % (TROW * 24);
  const int t  = r / 24;
  const int m3 = (r % 24) >> 3;   // 0=q 1=k 2=v
  const int h  = (r % 24) & 7;

  const bf16* arow = Xb + ((size_t)b * ROWP + t * 16 + (lane & 15)) * C_;
  const int cb = m3 * C_ + h * D_;

  v8f acc[4] = {};
  for (int K0 = 0; K0 < C_; K0 += 32) {
    v16bf a = loadA_bf(arow, K0, lane);
#pragma unroll
    for (int j = 0; j < 4; ++j)
      acc[j] = wmma_bf(a, loadB_nk(Wqkvt, C_, cb + j * 16, K0, lane), acc[j]);
  }

  if (m3 < 2) {  // L2-normalize q,k rows over D=64 (cols spread over 16 lanes x 4 j)
#pragma unroll
    for (int rr = 0; rr < 8; ++rr) {
      float s = 0.f;
#pragma unroll
      for (int j = 0; j < 4; ++j) s += acc[j][rr] * acc[j][rr];
      for (int off = 1; off < 16; off <<= 1) s += __shfl_xor(s, off, 16);
      float sc = 1.0f / fmaxf(sqrtf(s), 1e-12f);
#pragma unroll
      for (int j = 0; j < 4; ++j) acc[j][rr] *= sc;
    }
    bf16* dst = (m3 == 0) ? qb : kb;
    const size_t base = ((size_t)b * H_ + h) * ROWP;
#pragma unroll
    for (int j = 0; j < 4; ++j)
#pragma unroll
      for (int rr = 0; rr < 8; ++rr) {
        int row = t * 16 + rr + ((lane >= 16) ? 8 : 0);
        dst[(base + row) * D_ + j * 16 + (lane & 15)] = (bf16)acc[j][rr];
      }
  } else {       // v: store transposed [B][H][64][row]
    const size_t base = ((size_t)b * H_ + h) * D_;
#pragma unroll
    for (int j = 0; j < 4; ++j)
#pragma unroll
      for (int rr = 0; rr < 8; ++rr) {
        int row = t * 16 + rr + ((lane >= 16) ? 8 : 0);
        vT[(base + j * 16 + (lane & 15)) * ROWP + row] = (bf16)acc[j][rr];
      }
  }
}

// ---------------- Kernel 2: flash attention per (b, h, 16-row q tile) --------
__global__ __launch_bounds__(32) void attn_kernel(const bf16* __restrict__ qb,
                                                  const bf16* __restrict__ kb,
                                                  const bf16* __restrict__ vT,
                                                  const float* __restrict__ temperature,
                                                  bf16* __restrict__ Ob) {
  __shared__ bf16 lds_p[16 * 32];   // P tile staging: D-layout -> A-layout
  const int lane = threadIdx.x;
  int idx = blockIdx.x;
  const int b = idx / (H_ * QTR);
  const int r = idx % (H_ * QTR);
  const int h = r / QTR;
  const int t = r % QTR;
  const int q0 = t * 16;

  const size_t headrow = ((size_t)b * H_ + h) * ROWP;
  const bf16* qrow   = qb + (headrow + q0 + (lane & 15)) * D_;
  const bf16* vThead = vT + ((size_t)b * H_ + h) * D_ * ROWP;
  const v16bf aq0 = loadA_bf(qrow, 0,  lane);   // resident Q: K = d 0..31
  const v16bf aq1 = loadA_bf(qrow, 32, lane);   //             K = d 32..63
  const float tmp = temperature[h];

  float mrow[8], lrow[8];
  v8f o[4] = {};
#pragma unroll
  for (int rr = 0; rr < 8; ++rr) { mrow[rr] = -3.0e38f; lrow[rr] = 0.f; }

  for (int kv0 = 0; kv0 < ROWP; kv0 += 32) {
    const bf16* kbase = kb + (headrow + kv0) * D_;

    // S = (q . k^T) * temp : 16 x 32 in two 16x16 D-tiles (N=kv, K=d)
    v8f s[2] = {};
#pragma unroll
    for (int j = 0; j < 2; ++j) {
      s[j] = wmma_bf(aq0, loadB_nk(kbase, D_, j * 16, 0,  lane), s[j]);
      s[j] = wmma_bf(aq1, loadB_nk(kbase, D_, j * 16, 32, lane), s[j]);
    }
#pragma unroll
    for (int j = 0; j < 2; ++j) {
      const bool oob = (kv0 + j * 16 + (lane & 15)) >= NT;
#pragma unroll
      for (int rr = 0; rr < 8; ++rr)
        s[j][rr] = oob ? -3.0e38f : s[j][rr] * tmp;
    }

    // online softmax: per-row max/sum across the 16 lanes of each half
    float mnew[8], alpha[8];
#pragma unroll
    for (int rr = 0; rr < 8; ++rr) {
      float pm = fmaxf(s[0][rr], s[1][rr]);
      for (int off = 1; off < 16; off <<= 1) pm = fmaxf(pm, __shfl_xor(pm, off, 16));
      mnew[rr]  = fmaxf(mrow[rr], pm);
      alpha[rr] = __expf(mrow[rr] - mnew[rr]);
      mrow[rr]  = mnew[rr];
    }
#pragma unroll
    for (int rr = 0; rr < 8; ++rr) {
      float p0 = __expf(s[0][rr] - mnew[rr]);
      float p1 = __expf(s[1][rr] - mnew[rr]);
      s[0][rr] = p0; s[1][rr] = p1;
      float ps = p0 + p1;
      for (int off = 1; off < 16; off <<= 1) ps += __shfl_xor(ps, off, 16);
      lrow[rr] = lrow[rr] * alpha[rr] + ps;
    }
#pragma unroll
    for (int j = 0; j < 4; ++j)
#pragma unroll
      for (int rr = 0; rr < 8; ++rr) o[j][rr] *= alpha[rr];

    // P: D-layout regs -> LDS -> A-layout regs
    const int Mo = (lane >= 16) ? 8 : 0;
#pragma unroll
    for (int j = 0; j < 2; ++j)
#pragma unroll
      for (int rr = 0; rr < 8; ++rr)
        lds_p[(rr + Mo) * 32 + j * 16 + (lane & 15)] = (bf16)s[j][rr];
    __syncthreads();
    v16bf ap = loadA_bf(lds_p + (size_t)(lane & 15) * 32, 0, lane);
    __syncthreads();

    // O += P @ V   (B from v^T: N=d contiguous in kv)
#pragma unroll
    for (int j = 0; j < 4; ++j)
      o[j] = wmma_bf(ap, loadB_nk(vThead, ROWP, j * 16, kv0, lane), o[j]);
  }

  const size_t obase = (size_t)b * ROWP * C_;
#pragma unroll
  for (int rr = 0; rr < 8; ++rr) {
    int row = q0 + rr + ((lane >= 16) ? 8 : 0);
    if (row < NT) {
      float inv = 1.0f / lrow[rr];
#pragma unroll
      for (int j = 0; j < 4; ++j)
        Ob[obase + (size_t)row * C_ + h * D_ + j * 16 + (lane & 15)] =
            (bf16)(o[j][rr] * inv);
    }
  }
}

// ---------------- Kernel 3: output projection, first N rows only -------------
__global__ __launch_bounds__(32) void out_kernel(const bf16* __restrict__ Ob,
                                                 const bf16* __restrict__ Woutt,
                                                 float* __restrict__ out) {
  const int lane = threadIdx.x;
  int idx = blockIdx.x;
  const int b  = idx / (144 * 8);
  const int r  = idx % (144 * 8);
  const int t  = r / 8;
  const int ct = r % 8;

  const bf16* arow = Ob + ((size_t)b * ROWP + t * 16 + (lane & 15)) * C_;
  v8f acc[4] = {};
  for (int K0 = 0; K0 < C_; K0 += 32) {
    v16bf a = loadA_bf(arow, K0, lane);
#pragma unroll
    for (int j = 0; j < 4; ++j)
      acc[j] = wmma_bf(a, loadB_nk(Woutt, C_, ct * 64 + j * 16, K0, lane), acc[j]);
  }
#pragma unroll
  for (int j = 0; j < 4; ++j)
#pragma unroll
    for (int rr = 0; rr < 8; ++rr) {
      int row = t * 16 + rr + ((lane >= 16) ? 8 : 0);
      out[((size_t)b * N_ + row) * C_ + ct * 64 + j * 16 + (lane & 15)] = acc[j][rr];
    }
}

extern "C" void kernel_launch(void* const* d_in, const int* in_sizes, int n_in,
                              void* d_out, int out_size, void* d_ws, size_t ws_size,
                              hipStream_t stream) {
  const float* X    = (const float*)d_in[0];
  const float* S    = (const float*)d_in[1];
  const float* Wqkv = (const float*)d_in[2];
  const float* Wout = (const float*)d_in[3];
  const float* temp = (const float*)d_in[4];
  float* out = (float*)d_out;

  char* ws = (char*)d_ws;
  const size_t xin_e = (size_t)B_ * ROWP * C_;        // 4,784,128 elems
  const size_t qkv_e = (size_t)B_ * H_ * ROWP * D_;   // 4,784,128 elems
  const size_t wq_e  = (size_t)3 * C_ * C_;           // 786,432
  const size_t wo_e  = (size_t)C_ * C_;               // 262,144

  bf16* Xb    = (bf16*)(ws);                  ws += xin_e * sizeof(bf16);
  bf16* Wqkvt = (bf16*)(ws);                  ws += wq_e  * sizeof(bf16);
  bf16* Woutt = (bf16*)(ws);                  ws += wo_e  * sizeof(bf16);
  bf16* qb    = (bf16*)(ws);                  ws += qkv_e * sizeof(bf16);
  bf16* kb    = (bf16*)(ws);                  ws += qkv_e * sizeof(bf16);
  bf16* vT    = (bf16*)(ws);                  ws += qkv_e * sizeof(bf16);
  bf16* Ob    = (bf16*)(ws);                  // xin_e elems

  cvt_x_kernel<<<dim3((unsigned)(xin_e / 4 / 256)), dim3(256), 0, stream>>>(X, S, Xb);
  transpose_w_kernel<<<dim3((unsigned)(wq_e / 256)), dim3(256), 0, stream>>>(
      Wqkv, Wqkvt, C_, 3 * C_);
  transpose_w_kernel<<<dim3((unsigned)(wo_e / 256)), dim3(256), 0, stream>>>(
      Wout, Woutt, C_, C_);
  qkv_kernel<<<dim3(B_ * TROW * 24), dim3(32), 0, stream>>>(Xb, Wqkvt, qb, kb, vT);
  attn_kernel<<<dim3(B_ * H_ * QTR), dim3(32), 0, stream>>>(qb, kb, vT, temp, Ob);
  out_kernel <<<dim3(B_ * 144 * 8),  dim3(32), 0, stream>>>(Ob, Woutt, out);
}
